// MarginLoss_67662914781739
// MI455X (gfx1250) — compile-verified, matrix operand order checked by reading, not run on hardware
//
#include <hip/hip_runtime.h>

#define KROWS 8192
#define DDIM  768
#define NGRP  256
#define FINF  1e9f

typedef __attribute__((ext_vector_type(16))) __bf16 v16bf;
typedef __attribute__((ext_vector_type(8)))  __bf16 v8bf;
typedef __attribute__((ext_vector_type(8)))  float  v8f;

union Frag { v16bf v; v8bf h[2]; };

__device__ __forceinline__ unsigned short f2bf(float f) {
  unsigned u = __float_as_uint(f);
  u += 0x7FFFu + ((u >> 16) & 1u);   // round-to-nearest-even to bf16
  return (unsigned short)(u >> 16);
}

// Kernel 1: split fp32 X into bf16 hi + bf16 lo residual (x ~= hi + lo)
__global__ void __launch_bounds__(256) split_hi_lo(const float* __restrict__ x,
                                                   unsigned short* __restrict__ hi,
                                                   unsigned short* __restrict__ lo,
                                                   int n) {
  int i = blockIdx.x * blockDim.x + threadIdx.x;
  if (i < n) {
    float v = x[i];
    unsigned short h = f2bf(v);
    float hf = __uint_as_float(((unsigned)h) << 16);
    hi[i] = h;
    lo[i] = f2bf(v - hf);
  }
}

// Kernel 2: fused Gram tile GEMM with masked row min/max epilogue.
// 32-row block per workgroup (two 16-row sub-tiles per wave -> bh/bl pair
// feeds 6 WMMAs). A block staged in dynamic LDS (96 KB) and re-read per chunk.
// Six independent accumulators break the WMMA RAW chain; unrolling constrained
// to keep ~164 VGPRs (no spills). Epilogue is branchless (select + min/max).
__global__ void __launch_bounds__(256) gram_rowminmax(const unsigned short* __restrict__ Xhi,
                                                      const unsigned short* __restrict__ Xlo,
                                                      const int* __restrict__ pair,
                                                      float* __restrict__ row_min,
                                                      float* __restrict__ row_max) {
  extern __shared__ __align__(16) unsigned short smem[];
  unsigned short* Ah = smem;                 // 32*DDIM bf16 = 48 KB
  unsigned short* Al = smem + 32 * DDIM;     // 32*DDIM bf16 = 48 KB
  __shared__ float wmn[8][32];
  __shared__ float wmx[8][32];

  const int tid  = threadIdx.x;
  const int wave = tid >> 5;
  const int lane = tid & 31;
  const int rowBase = blockIdx.x * 32;

  { // stage 32 x DDIM hi/lo rows into LDS, 16B per transfer
    const uint4* gh = (const uint4*)(Xhi + (size_t)rowBase * DDIM);
    const uint4* gl = (const uint4*)(Xlo + (size_t)rowBase * DDIM);
    uint4* dh = (uint4*)Ah;
    uint4* dl = (uint4*)Al;
    for (int t = tid; t < 32 * DDIM / 8; t += 256) { dh[t] = gh[t]; dl[t] = gl[t]; }
  }
  __syncthreads();

  const int half = lane >> 4;     // which 16-lane half
  const int l16  = lane & 15;
  const int koff = half * 8;      // K sub-offset per half (16-bit operand layout)

  // per-lane LDS row bases (sub-tile 0 and 1), hoisted
  const unsigned short* Ah0 = Ah + l16 * DDIM;
  const unsigned short* Al0 = Al + l16 * DDIM;
  const unsigned short* Ah1 = Ah + (16 + l16) * DDIM;
  const unsigned short* Al1 = Al + (16 + l16) * DDIM;

  int gi[2][8];
#pragma unroll
  for (int s = 0; s < 2; ++s)
#pragma unroll
    for (int v = 0; v < 8; ++v)
      gi[s][v] = pair[2 * (rowBase + s * 16 + v + half * 8) + 1];

  float rmn[2][8], rmx[2][8];
#pragma unroll
  for (int s = 0; s < 2; ++s)
#pragma unroll
    for (int v = 0; v < 8; ++v) { rmn[s][v] = FINF; rmx[s][v] = -FINF; }

#pragma unroll 1
  for (int ct = wave; ct < KROWS / 16; ct += 8) {
    const int colBase = ct * 16;
    const int j  = colBase + l16;
    const int gj = pair[2 * j + 1];
    const unsigned short* bh_row = Xhi + (size_t)j * DDIM;
    const unsigned short* bl_row = Xlo + (size_t)j * DDIM;

    // independent accumulators: {hi*hi, hi*lo, lo*hi} x {subtile0, subtile1}
    v8f hh0 = {}, hl0 = {}, lh0 = {};
    v8f hh1 = {}, hl1 = {}, lh1 = {};

#pragma unroll 2
    for (int ch = 0; ch < DDIM / 32; ++ch) {
      const int k0 = ch * 32 + koff;
      Frag ah0, al0, ah1, al1, bh, bl;
      // elements 0..7 <- k0..k0+7 ; elements 8..15 <- k0+16..k0+23
      bh.h[0]  = *(const v8bf*)&bh_row[k0];
      bh.h[1]  = *(const v8bf*)&bh_row[k0 + 16];
      bl.h[0]  = *(const v8bf*)&bl_row[k0];
      bl.h[1]  = *(const v8bf*)&bl_row[k0 + 16];
      ah0.h[0] = *(const v8bf*)&Ah0[k0];
      ah0.h[1] = *(const v8bf*)&Ah0[k0 + 16];
      al0.h[0] = *(const v8bf*)&Al0[k0];
      al0.h[1] = *(const v8bf*)&Al0[k0 + 16];
      ah1.h[0] = *(const v8bf*)&Ah1[k0];
      ah1.h[1] = *(const v8bf*)&Ah1[k0 + 16];
      al1.h[0] = *(const v8bf*)&Al1[k0];
      al1.h[1] = *(const v8bf*)&Al1[k0 + 16];

      hh0 = __builtin_amdgcn_wmma_f32_16x16x32_bf16(false, ah0.v, false, bh.v, (short)0, hh0, false, false);
      hl0 = __builtin_amdgcn_wmma_f32_16x16x32_bf16(false, ah0.v, false, bl.v, (short)0, hl0, false, false);
      lh0 = __builtin_amdgcn_wmma_f32_16x16x32_bf16(false, al0.v, false, bh.v, (short)0, lh0, false, false);
      hh1 = __builtin_amdgcn_wmma_f32_16x16x32_bf16(false, ah1.v, false, bh.v, (short)0, hh1, false, false);
      hl1 = __builtin_amdgcn_wmma_f32_16x16x32_bf16(false, ah1.v, false, bl.v, (short)0, hl1, false, false);
      lh1 = __builtin_amdgcn_wmma_f32_16x16x32_bf16(false, al1.v, false, bh.v, (short)0, lh1, false, false);
    }

    const v8f c0 = hh0 + hl0 + lh0;
    const v8f c1 = hh1 + hl1 + lh1;

    // Branchless masked epilogue (select sentinel, then min/max; matches
    // reference's where(..., S, +/-INF) semantics). C layout: lane holds
    // column N = l16; vgpr v holds row M = v + 8*half (sub-tile 1 adds +16).
#pragma unroll
    for (int v = 0; v < 8; ++v) {
      {
        const int  i    = rowBase + v + half * 8;
        const float s   = c0[v];
        const bool same = (gi[0][v] == gj);
        const float cmn = (same && (i != j)) ? s : FINF;
        const float cmx = same ? -FINF : s;
        rmn[0][v] = fminf(rmn[0][v], cmn);
        rmx[0][v] = fmaxf(rmx[0][v], cmx);
      }
      {
        const int  i    = rowBase + 16 + v + half * 8;
        const float s   = c1[v];
        const bool same = (gi[1][v] == gj);
        const float cmn = (same && (i != j)) ? s : FINF;
        const float cmx = same ? -FINF : s;
        rmn[1][v] = fminf(rmn[1][v], cmn);
        rmx[1][v] = fmaxf(rmx[1][v], cmx);
      }
    }
  }

  // Reduce across the 16 lanes sharing each row (xor<16 keeps halves separate)
#pragma unroll
  for (int s = 0; s < 2; ++s)
#pragma unroll
    for (int v = 0; v < 8; ++v) {
#pragma unroll
      for (int off = 1; off < 16; off <<= 1) {
        rmn[s][v] = fminf(rmn[s][v], __shfl_xor(rmn[s][v], off, 32));
        rmx[s][v] = fmaxf(rmx[s][v], __shfl_xor(rmx[s][v], off, 32));
      }
    }

  if (l16 == 0) {   // lanes 0 (rows +0..7) and 16 (rows +8..15)
#pragma unroll
    for (int s = 0; s < 2; ++s)
#pragma unroll
      for (int v = 0; v < 8; ++v) {
        wmn[wave][s * 16 + v + half * 8] = rmn[s][v];
        wmx[wave][s * 16 + v + half * 8] = rmx[s][v];
      }
  }
  __syncthreads();

  if (tid < 32) {
    float m = FINF, M = -FINF;
#pragma unroll
    for (int w = 0; w < 8; ++w) { m = fminf(m, wmn[w][tid]); M = fmaxf(M, wmx[w][tid]); }
    row_min[rowBase + tid] = m;
    row_max[rowBase + tid] = M;
  }
}

// Order-preserving float <-> uint keys so LDS integer atomics do float min/max
__device__ __forceinline__ unsigned fkey(float f) {
  unsigned u = __float_as_uint(f);
  return (u & 0x80000000u) ? ~u : (u | 0x80000000u);
}
__device__ __forceinline__ float unkey(unsigned k) {
  unsigned u = (k & 0x80000000u) ? (k ^ 0x80000000u) : ~k;
  return __uint_as_float(u);
}

// Kernel 3: segment min/max over 256 groups + final margin loss scalar
__global__ void __launch_bounds__(256) finalize_loss(const float* __restrict__ row_min,
                                                     const float* __restrict__ row_max,
                                                     const int* __restrict__ pair,
                                                     float* __restrict__ out) {
  __shared__ unsigned gmin[NGRP];
  __shared__ unsigned gmax[NGRP];
  __shared__ int      gcnt[NGRP];
  __shared__ float    rsum[256];
  __shared__ int      rcnt[256];
  const int tid = threadIdx.x;
  gmin[tid] = 0xFFFFFFFFu; gmax[tid] = 0u; gcnt[tid] = 0;
  __syncthreads();
  for (int i = tid; i < KROWS; i += 256) {
    int grp = pair[2 * i + 1];
    atomicMin(&gmin[grp], fkey(row_min[i]));
    atomicMax(&gmax[grp], fkey(row_max[i]));
    atomicAdd(&gcnt[grp], 1);
  }
  __syncthreads();
  float per = 0.f; int ne = 0;
  if (gcnt[tid] > 0) {
    ne = 1;
    per = fmaxf(0.f, unkey(gmax[tid]) - unkey(gmin[tid]) + 0.1f);
  }
  rsum[tid] = per; rcnt[tid] = ne;
  __syncthreads();
  for (int s = 128; s > 0; s >>= 1) {
    if (tid < s) { rsum[tid] += rsum[tid + s]; rcnt[tid] += rcnt[tid + s]; }
    __syncthreads();
  }
  if (tid == 0) out[0] = rsum[0] / (float)rcnt[0];
}

extern "C" void kernel_launch(void* const* d_in, const int* in_sizes, int n_in,
                              void* d_out, int out_size, void* d_ws, size_t ws_size,
                              hipStream_t stream) {
  (void)in_sizes; (void)n_in; (void)out_size; (void)ws_size;
  const float* X  = (const float*)d_in[0];
  const int* pair = (const int*)d_in[1];   // pair_idx canonicalized to int32; g = pair[2*k+1]
  float* out = (float*)d_out;

  char* w = (char*)d_ws;
  const size_t nElem = (size_t)KROWS * DDIM;
  unsigned short* Xhi = (unsigned short*)w;                       // 12.58 MB
  unsigned short* Xlo = (unsigned short*)(w + nElem * 2);         // 12.58 MB
  float* rmn = (float*)(w + nElem * 4);                           // 32 KB
  float* rmx = (float*)(w + nElem * 4 + (size_t)KROWS * 4);       // 32 KB

  split_hi_lo<<<(int)((nElem + 255) / 256), 256, 0, stream>>>(X, Xhi, Xlo, (int)nElem);

  const size_t ldsBytes = (size_t)2 * 32 * DDIM * sizeof(unsigned short);  // 96 KB
  gram_rowminmax<<<KROWS / 32, 256, ldsBytes, stream>>>(Xhi, Xlo, pair, rmn, rmx);

  finalize_loss<<<1, 256, 0, stream>>>(rmn, rmx, pair, out);
}
